// AdditiveAttentionLayer_46712064311859
// MI455X (gfx1250) — matrix operand order repaired and checked by reading, not compiled, not run on hardware
//
#include <hip/hip_runtime.h>
#include <stdint.h>

#define TT 1024
#define BB 32
#define HH 1024
#define NROWS (TT * BB)       // 32768 rows, row = t*BB + b

typedef __attribute__((ext_vector_type(16))) __bf16 v16bf;
typedef __attribute__((ext_vector_type(8)))  float  v8f;

union Frag {
    uint4  q[2];   // two 16-byte loads
    v16bf  v;      // 16 bf16 = 8 VGPRs, WMMA A/B operand
};

__device__ __forceinline__ uint32_t pk_bf16(float a, float b) {
    uint32_t ua = __float_as_uint(a);
    uint32_t ub = __float_as_uint(b);
    ua = (ua + 0x7FFFu + ((ua >> 16) & 1u)) >> 16;   // round-to-nearest-even
    ub = (ub + 0x7FFFu + ((ub >> 16) & 1u)) >> 16;
    return ua | (ub << 16);
}

// ---------------------------------------------------------------------------
// Kernel 0a: Wt_bf16[n][k] = bf16(W[k][n])   (transpose so B-frags are
// contiguous-per-lane in the K dimension). 1024x1024.
// ---------------------------------------------------------------------------
__global__ __launch_bounds__(256) void k_cvtW(const float* __restrict__ W,
                                              uint32_t* __restrict__ Wt) {
    const int idx = blockIdx.x * 256 + threadIdx.x;   // 0 .. 1024*512-1
    const int n  = idx >> 9;
    const int k2 = idx & 511;
    const int k  = k2 << 1;
    const float a = W[(size_t)k * HH + n];
    const float b = W[(size_t)(k + 1) * HH + n];
    Wt[(size_t)n * 512 + k2] = pk_bf16(a, b);
}

// ---------------------------------------------------------------------------
// Kernel 0b: CW_bf16[n][k] = bf16(concat_w[n][k])  (torch Linear is [out,in],
// which is already the B^T layout we want). 1024x2048.
// ---------------------------------------------------------------------------
__global__ __launch_bounds__(256) void k_cvtCW(const float* __restrict__ CWf,
                                               uint32_t* __restrict__ CW) {
    const int idx = blockIdx.x * 256 + threadIdx.x;   // 0 .. 1024*1024-1
    const float2 f = *(const float2*)(CWf + (size_t)idx * 2);
    CW[idx] = pk_bf16(f.x, f.y);
}

// ---------------------------------------------------------------------------
// Kernel 1: scores[row] = tanh(X[row,:] @ W) . proj   via bf16 WMMA.
// Block = 16 rows, 8 waves; wave handles 8 N-tiles of 16 columns.
// A tile (16 x 1024 bf16) staged in LDS, row stride 2064 B (conflict-free
// ds_load_b128: stride/4 mod 64 == 4).
// ---------------------------------------------------------------------------
#define RB1 2064
__global__ __launch_bounds__(256) void k_scores(const float* __restrict__ X,
                                                const uint16_t* __restrict__ Wt,
                                                const float* __restrict__ proj,
                                                float* __restrict__ scores) {
    extern __shared__ unsigned char lds[];
    float* scRed = (float*)(lds + 16 * RB1);
    const int tid = threadIdx.x;
    const int rb  = blockIdx.x * 16;

    // Stage A tile: 16 rows x 1024 f32 -> bf16 in LDS
    for (int p = tid; p < 16 * 512; p += 256) {
        const int m = p >> 9;
        const int c = (p & 511) << 1;
        const float2 f = *(const float2*)(X + (size_t)(rb + m) * HH + c);
        *(uint32_t*)(lds + m * RB1 + (c << 1)) = pk_bf16(f.x, f.y);
    }
    if (tid < 16) scRed[tid] = 0.0f;
    __syncthreads();

    const int lane = tid & 31;
    const int l16  = lane & 15;
    const int half = lane >> 4;
    const int wave = tid >> 5;

    v8f acc[8];
    const v8f vzero = {0.f, 0.f, 0.f, 0.f, 0.f, 0.f, 0.f, 0.f};
#pragma unroll
    for (int j = 0; j < 8; ++j) acc[j] = vzero;

    for (int kc = 0; kc < 32; ++kc) {
        const int k0 = kc << 5;
        Frag a;
        a.q[0] = *(const uint4*)(lds + l16 * RB1 + ((k0 + (half << 3)) << 1));
        a.q[1] = *(const uint4*)(lds + l16 * RB1 + ((k0 + 16 + (half << 3)) << 1));
#pragma unroll
        for (int j = 0; j < 8; ++j) {
            const int n0 = ((wave << 3) + j) << 4;
            const uint16_t* bp = Wt + (size_t)(n0 + l16) * HH + k0 + (half << 3);
            Frag b;
            b.q[0] = *(const uint4*)bp;
            b.q[1] = *(const uint4*)(bp + 16);
            acc[j] = __builtin_amdgcn_wmma_f32_16x16x32_bf16(
                false, a.v, false, b.v, (short)0, acc[j], false, false);
        }
    }

    // score partials: row m = r + 8*half, column n = n0 + l16
    float pr[8];
#pragma unroll
    for (int r = 0; r < 8; ++r) pr[r] = 0.f;
#pragma unroll
    for (int j = 0; j < 8; ++j) {
        const int n0 = ((wave << 3) + j) << 4;
        const float pj = proj[n0 + l16];
#pragma unroll
        for (int r = 0; r < 8; ++r) pr[r] += tanhf(acc[j][r]) * pj;
    }
    // reduce across the 16 lanes of each half (xor of bits 0..3 stays in-half)
#pragma unroll
    for (int mask = 1; mask <= 8; mask <<= 1) {
#pragma unroll
        for (int r = 0; r < 8; ++r) pr[r] += __shfl_xor(pr[r], mask, 32);
    }
    if (l16 == 0) {
#pragma unroll
        for (int r = 0; r < 8; ++r) atomicAdd(&scRed[r + (half << 3)], pr[r]);
    }
    __syncthreads();
    if (tid < 16) scores[rb + tid] = scRed[tid];
}

// ---------------------------------------------------------------------------
// Kernel 2: online-softmax prefix scan -> context, written into io (= d_out)
// for rows t >= 2. One block per batch b; thread owns 4 consecutive h.
// The 4 KB scores column is a pure f32 copy to LDS -> use CDNA5 async
// memory->LDS loads (ASYNCcnt) instead of a VGPR round-trip.
// ---------------------------------------------------------------------------
__global__ __launch_bounds__(256) void k_ctx(const float* __restrict__ X,
                                             const float* __restrict__ scores,
                                             float* __restrict__ io) {
    __shared__ float sc[TT];
    const int b   = blockIdx.x;
    const int tid = threadIdx.x;

    // async copy: scores[s*BB + b] -> sc[s], s = tid, tid+256, ...
#pragma unroll
    for (int i = 0; i < TT / 256; ++i) {
        const int s = tid + i * 256;
        const uint64_t ga = (uint64_t)(const void*)(scores + (size_t)s * BB + b);
        const uint32_t la = (uint32_t)(uintptr_t)(&sc[s]);
        asm volatile("global_load_async_to_lds_b32 %0, %1, off"
                     :: "v"(la), "v"(ga) : "memory");
    }
    asm volatile("s_wait_asynccnt 0x0" ::: "memory");
    __syncthreads();

    const int h = tid << 2;
    float nx = 0.f, ny = 0.f, nz = 0.f, nw = 0.f;
    float den  = 0.f;
    float mrun = -__builtin_inff();

    for (int t0 = 0; t0 < TT; t0 += 4) {
        float4 xv[4];
        float  sv[4];
#pragma unroll
        for (int j = 0; j < 4; ++j) {   // batch independent loads to hide latency
            xv[j] = *(const float4*)(X + ((size_t)(t0 + j) * BB + b) * HH + h);
            sv[j] = sc[t0 + j];
        }
#pragma unroll
        for (int j = 0; j < 4; ++j) {
            const int t = t0 + j;
            if (t >= 2) {               // strict prefix: emit before absorbing s=t
                const float inv = 1.0f / den;
                float4 c;
                c.x = nx * inv; c.y = ny * inv; c.z = nz * inv; c.w = nw * inv;
                *(float4*)(io + ((size_t)t * BB + b) * HH + h) = c;
            }
            const float mn = fmaxf(mrun, sv[j]);
            const float f  = expf(mrun - mn);
            const float e  = expf(sv[j] - mn);
            nx = nx * f + e * xv[j].x;
            ny = ny * f + e * xv[j].y;
            nz = nz * f + e * xv[j].z;
            nw = nw * f + e * xv[j].w;
            den  = den * f + e;
            mrun = mn;
        }
    }
}

// ---------------------------------------------------------------------------
// Kernel 3: out = tanh([X | context] @ concat_w^T + b), in place in io.
// Block = 16 rows; cat (16 x 2048) staged as bf16 in LDS (stride 4112 B,
// conflict-free). Rows with t < 2 pass inputs through.
// ---------------------------------------------------------------------------
#define RB3 4112
__global__ __launch_bounds__(256) void k_out(const float* __restrict__ X,
                                             const uint16_t* __restrict__ CW,
                                             const float* __restrict__ Cb,
                                             float* io) {
    extern __shared__ unsigned char lds[];
    const int tid = threadIdx.x;
    const int rb  = blockIdx.x * 16;

    // Stage cat tile: [inputs(0:1024) | context(1024:2048)] -> bf16 LDS
    for (int p = tid; p < 16 * 1024; p += 256) {
        const int m = p >> 10;
        const int c = (p & 1023) << 1;
        const float* src = (c < 1024)
            ? (X  + (size_t)(rb + m) * HH + c)
            : (io + (size_t)(rb + m) * HH + (c - 1024));
        const float2 f = *(const float2*)src;
        *(uint32_t*)(lds + m * RB3 + (c << 1)) = pk_bf16(f.x, f.y);
    }
    __syncthreads();

    const int lane = tid & 31;
    const int l16  = lane & 15;
    const int half = lane >> 4;
    const int wave = tid >> 5;

    v8f acc[8];
    const v8f vzero = {0.f, 0.f, 0.f, 0.f, 0.f, 0.f, 0.f, 0.f};
#pragma unroll
    for (int j = 0; j < 8; ++j) acc[j] = vzero;

    for (int kc = 0; kc < 64; ++kc) {
        const int k0 = kc << 5;
        Frag a;
        a.q[0] = *(const uint4*)(lds + l16 * RB3 + ((k0 + (half << 3)) << 1));
        a.q[1] = *(const uint4*)(lds + l16 * RB3 + ((k0 + 16 + (half << 3)) << 1));
#pragma unroll
        for (int j = 0; j < 8; ++j) {
            const int n0 = ((wave << 3) + j) << 4;
            const uint16_t* bp = CW + (size_t)(n0 + l16) * 2048 + k0 + (half << 3);
            Frag b;
            b.q[0] = *(const uint4*)bp;
            b.q[1] = *(const uint4*)(bp + 16);
            __builtin_prefetch((const void*)(bp + 32), 0, 0);  // next K chunk
            acc[j] = __builtin_amdgcn_wmma_f32_16x16x32_bf16(
                false, a.v, false, b.v, (short)0, acc[j], false, false);
        }
    }

#pragma unroll
    for (int j = 0; j < 8; ++j) {
        const int n  = (((wave << 3) + j) << 4) + l16;
        const float bv = Cb[n];
#pragma unroll
        for (int r = 0; r < 8; ++r) {
            const int g = rb + r + (half << 3);   // global row = t*BB + b
            const int t = g >> 5;
            float v = tanhf(acc[j][r] + bv);
            if (t < 2) v = X[(size_t)g * HH + n]; // steps 0,1 pass through
            io[(size_t)g * HH + n] = v;
        }
    }
}

// ---------------------------------------------------------------------------
extern "C" void kernel_launch(void* const* d_in, const int* in_sizes, int n_in,
                              void* d_out, int out_size, void* d_ws, size_t ws_size,
                              hipStream_t stream) {
    (void)in_sizes; (void)n_in; (void)out_size; (void)ws_size;

    const float* X    = (const float*)d_in[0];   // [T,B,H]
    const float* W    = (const float*)d_in[1];   // [H,H]
    const float* proj = (const float*)d_in[2];   // [H]
    const float* CWf  = (const float*)d_in[3];   // [H,2H]
    const float* Cb   = (const float*)d_in[4];   // [H]
    float* out = (float*)d_out;

    // ws layout: scores (128 KB) | Wt bf16 (2 MB) | CW bf16 (4 MB)
    float*    scores = (float*)d_ws;
    uint32_t* Wt32   = (uint32_t*)((char*)d_ws + (size_t)NROWS * 4);
    uint32_t* CW32   = (uint32_t*)((char*)d_ws + (size_t)NROWS * 4 + (size_t)HH * HH * 2);

    k_cvtW <<<2048, 256, 0, stream>>>(W, Wt32);
    k_cvtCW<<<4096, 256, 0, stream>>>(CWf, CW32);
    k_scores<<<NROWS / 16, 256, 16 * RB1 + 64, stream>>>(
        X, (const uint16_t*)Wt32, proj, scores);
    k_ctx<<<BB, 256, 0, stream>>>(X, scores, out);
    k_out<<<NROWS / 16, 256, 16 * RB3, stream>>>(
        X, (const uint16_t*)CW32, Cb, out);
}
